// AttentionAggregation_56951266345225
// MI455X (gfx1250) — matrix operand order repaired and checked by reading, not compiled, not run on hardware
//
#include <hip/hip_runtime.h>

typedef __attribute__((ext_vector_type(16))) _Float16 v16h;
typedef __attribute__((ext_vector_type(8)))  float    v8f;

#define BDIM 2
#define FDIM 4
#define NQ   4096
#define MQ   8192
#define CC   64
#define HS   256
#define WS   256
#define HWC  (HS * WS)

// ---------------------------------------------------------------------------
// Scatter previous-frame sparse features into the dense BEV grid in d_ws.
// One float4 (4 channels) per thread; rows are unique per bucket (no races).
// ---------------------------------------------------------------------------
__global__ void scatter_pre_kernel(const float* __restrict__ pre,
                                   const int* __restrict__ pidx,
                                   float* __restrict__ dense) {
  int t = blockIdx.x * blockDim.x + threadIdx.x;
  const int total = BDIM * FDIM * MQ * (CC / 4);
  if (t >= total) return;
  int r  = t / (CC / 4);
  int c4 = t % (CC / 4);
  const int* row = pidx + (size_t)r * 4;
  int bf  = row[0] * FDIM + (row[1] - 1);
  int pos = row[2] * WS + row[3];
  const float4* src = (const float4*)(pre + (size_t)r * CC);
  float4* dst = (float4*)(dense + ((size_t)bf * HWC + pos) * CC);
  dst[c4] = src[c4];
}

// ---------------------------------------------------------------------------
// One-wave prep kernel: pack the stage weights [We | Ww] (64 x 16, padded)
// into per-lane f16 WMMA B-fragments (contiguous k = 16*half + s per lane) plus the
// bias vector for the C accumulator. Done once; every tile block then loads
// two 32-byte fragments instead of re-gathering weights with branchy code.
// bfrag layout: [stage][lane][2] v16h ; bias layout: [stage][16] float.
// ---------------------------------------------------------------------------
__global__ __launch_bounds__(32) void pack_weights_kernel(
    const float* __restrict__ se1w, const float* __restrict__ se1b,
    const float* __restrict__ sw1w, const float* __restrict__ sw1b,
    const float* __restrict__ se2w, const float* __restrict__ se2b,
    const float* __restrict__ sw2w, const float* __restrict__ sw2b,
    const float* __restrict__ se3w, const float* __restrict__ se3b,
    const float* __restrict__ sw3w, const float* __restrict__ sw3b,
    v16h* __restrict__ bfrag, float* __restrict__ bias) {
  int lane = threadIdx.x;
  int m    = lane & 15;   // B column
  int half = lane >> 4;
  const float* WeT[3] = {se1w, se2w, se3w};
  const float* beT[3] = {se1b, se2b, se3b};
  const float* WwT[3] = {sw1w, sw2w, sw3w};
  const float* bwT[3] = {sw1b, sw2b, sw3b};
  const int    pT[3]  = {4, 2, 1};
#pragma unroll
  for (int s = 0; s < 3; ++s) {
    int p = pT[s];
    const float* We = WeT[s];
    const float* Ww = WwT[s];
    v16h b0, b1;
#pragma unroll
    for (int v = 0; v < 8; ++v) {
      int kB = 16 * half + 2 * v;
#pragma unroll
      for (int j = 0; j < 2; ++j) {
        int kb0 = kB + j, kb1 = kb0 + 32;
        float w0 = (m < 2 * p) ? We[kb0 * 2 * p + m]
                 : ((m < 3 * p) ? Ww[kb0 * p + (m - 2 * p)] : 0.0f);
        float w1 = (m < 2 * p) ? We[kb1 * 2 * p + m]
                 : ((m < 3 * p) ? Ww[kb1 * p + (m - 2 * p)] : 0.0f);
        b0[2 * v + j] = (_Float16)w0;
        b1[2 * v + j] = (_Float16)w1;
      }
    }
    bfrag[(s * 32 + lane) * 2 + 0] = b0;
    bfrag[(s * 32 + lane) * 2 + 1] = b1;
    if (half == 0) {
      bias[s * 16 + m] = (m < 2 * p) ? beT[s][m]
                       : ((m < 3 * p) ? bwT[s][m - 2 * p] : 0.0f);
    }
  }
}

// ---------------------------------------------------------------------------
// One stage of the deformable-attention pipeline for a 16-query tile,
// executed by a single wave32. P = number of sampling points (4, 2, 1).
// refy/refx/sy/sx live in registers of lanes 0..15 (lane q owns query q).
// ---------------------------------------------------------------------------
template<int P>
__device__ __forceinline__ void run_stage(
    const v16h* __restrict__ bfragS, const float* __restrict__ biasS,
    const float* __restrict__ dense, size_t denseBase, int lane,
    float refy, float refx, float& sy, float& sx,
    float (*xt)[CC], float (*Dt)[16],
    float (*cw)[4][4], int (*cell)[4][4], float (*wts)[4]) {

  const int m    = lane & 15;   // A row / B,D column
  const int half = lane >> 4;

  // ---- A fragments from the LDS x tile; B fragments preloaded -------------
  float cval = biasS[m];
  v8f c;
#pragma unroll
  for (int r = 0; r < 8; ++r) c[r] = cval;

  v16h b0 = bfragS[lane * 2 + 0];
  v16h b1 = bfragS[lane * 2 + 1];
  v16h a0, a1;
#pragma unroll
  for (int v = 0; v < 8; ++v) {
    int kA = 16 * half + 2 * v;
#pragma unroll
    for (int j = 0; j < 2; ++j) {
      a0[2 * v + j] = (_Float16)xt[m][kA + j];
      a1[2 * v + j] = (_Float16)xt[m][kA + j + 32];
    }
  }

  // D(16x16) = x(16x64) * [We|Ww](64x16) + bias, via two K=32 WMMAs.
  c = __builtin_amdgcn_wmma_f32_16x16x32_f16(false, a0, false, b0,
                                             (short)0, c, false, false);
  c = __builtin_amdgcn_wmma_f32_16x16x32_f16(false, a1, false, b1,
                                             (short)0, c, false, false);

#pragma unroll
  for (int r = 0; r < 8; ++r) Dt[r + 8 * half][m] = c[r];
  __syncthreads();

  // ---- Per-query: softmax over P logits, argmax, corner setup -------------
  if (lane < 16) {
    int q = lane;
    float mx = -1e30f;
#pragma unroll
    for (int pt = 0; pt < P; ++pt) mx = fmaxf(mx, Dt[q][2 * P + pt]);
    float sum = 0.0f, e[P];
#pragma unroll
    for (int pt = 0; pt < P; ++pt) { e[pt] = __expf(Dt[q][2 * P + pt] - mx); sum += e[pt]; }
    int amax = 0; float bestl = Dt[q][2 * P];
#pragma unroll
    for (int pt = 1; pt < P; ++pt)
      if (Dt[q][2 * P + pt] > bestl) { bestl = Dt[q][2 * P + pt]; amax = pt; }

    float rs = 1.0f / sum;
#pragma unroll
    for (int pt = 0; pt < P; ++pt) {
      float dy = Dt[q][2 * pt], dx = Dt[q][2 * pt + 1];
      // offset uses pre-update speed (reference semantics)
      float ly = refy + (sy + dy) * (0.1f / (float)HS);
      float lx = refx + (sx + dx) * (0.1f / (float)WS);
      float yy = ly * (float)HS - 0.5f;
      float xx = lx * (float)WS - 0.5f;
      float y0f = floorf(yy), x0f = floorf(xx);
      float fy = yy - y0f, fx = xx - x0f;
      int y0 = (int)y0f, x0 = (int)x0f;
      float wc[4] = {(1.f - fy) * (1.f - fx), (1.f - fy) * fx,
                     fy * (1.f - fx),         fy * fx};
      const int dyc[4] = {0, 0, 1, 1};
      const int dxc[4] = {0, 1, 0, 1};
#pragma unroll
      for (int cr = 0; cr < 4; ++cr) {
        int yi = y0 + dyc[cr], xi = x0 + dxc[cr];
        bool valid = (yi >= 0) & (yi < HS) & (xi >= 0) & (xi < WS);
        int yc = min(max(yi, 0), HS - 1);
        int xc = min(max(xi, 0), WS - 1);
        cell[q][pt][cr] = yc * WS + xc;
        cw[q][pt][cr]   = valid ? wc[cr] : 0.0f;
      }
      wts[q][pt] = e[pt] * rs;
    }
    sy += Dt[q][2 * amax];
    sx += Dt[q][2 * amax + 1];
  }
  __syncthreads();

  // ---- Weighted 4-corner gather: each lane owns 2 channels ----------------
  int ch = 2 * lane;
  for (int q = 0; q < 16; ++q) {
    float accx = 0.0f, accy = 0.0f;
#pragma unroll
    for (int pt = 0; pt < P; ++pt) {
      float gx = 0.0f, gy = 0.0f;
#pragma unroll
      for (int cr = 0; cr < 4; ++cr) {
        float w = cw[q][pt][cr];
        const float2 vv = *(const float2*)(dense + denseBase +
                                           (size_t)cell[q][pt][cr] * CC + ch);
        gx += w * vv.x;
        gy += w * vv.y;
      }
      float wp = wts[q][pt];
      accx += wp * gx;
      accy += wp * gy;
    }
    xt[q][ch]     = accx;
    xt[q][ch + 1] = accy;
  }
  __syncthreads();
}

// ---------------------------------------------------------------------------
// Fused 3-stage kernel: one wave32 per 16-query tile.
// ---------------------------------------------------------------------------
__global__ __launch_bounds__(32) void deform_stages_kernel(
    const float* __restrict__ cur, const float* __restrict__ dense,
    const int* __restrict__ cidx,
    const v16h* __restrict__ bfrag, const float* __restrict__ bias,
    float* __restrict__ outx, float* __restrict__ outspeed) {

  __shared__ float xt[16][CC];
  __shared__ float Dt[16][16];
  __shared__ float cw[16][4][4];
  __shared__ int   cell[16][4][4];
  __shared__ float wts[16][4];

  int lane = threadIdx.x;
  int q0   = blockIdx.x * 16;
  int bf   = q0 / NQ;  // whole tile lives in one (b,f) bucket (NQ % 16 == 0)
  size_t denseBase = (size_t)bf * HWC * CC;

  // Load the 16x64 query-feature tile into LDS (2 channels per lane).
  for (int q = 0; q < 16; ++q) {
    float2 v = *(const float2*)(cur + (size_t)(q0 + q) * CC + 2 * lane);
    xt[q][2 * lane]     = v.x;
    xt[q][2 * lane + 1] = v.y;
  }

  // Lane q (< 16) owns query q's reference location and speed state.
  float refy = 0.f, refx = 0.f, sy = 0.f, sx = 0.f;
  if (lane < 16) {
    const int* ci = cidx + (size_t)(q0 + lane) * 4;
    refy = (float)ci[2] / (float)HS;
    refx = (float)ci[3] / (float)WS;
  }
  __syncthreads();

  run_stage<4>(bfrag +   0, bias +  0, dense, denseBase, lane,
               refy, refx, sy, sx, xt, Dt, cw, cell, wts);
  run_stage<2>(bfrag +  64, bias + 16, dense, denseBase, lane,
               refy, refx, sy, sx, xt, Dt, cw, cell, wts);
  run_stage<1>(bfrag + 128, bias + 32, dense, denseBase, lane,
               refy, refx, sy, sx, xt, Dt, cw, cell, wts);

  // Write cur_out (natural order).
  for (int q = 0; q < 16; ++q) {
    float2 v = make_float2(xt[q][2 * lane], xt[q][2 * lane + 1]);
    *(float2*)(outx + (size_t)(q0 + q) * CC + 2 * lane) = v;
  }
  // Write speed with the reference's (f, b, n) reordering.
  if (lane < 16) {
    int qg = q0 + lane;
    int n  = qg % NQ;
    int b  = bf / FDIM, f = bf % FDIM;
    size_t orow = ((size_t)(f * BDIM + b) * NQ + n);
    outspeed[orow * 2]     = sy;
    outspeed[orow * 2 + 1] = sx;
  }
}

// ---------------------------------------------------------------------------
extern "C" void kernel_launch(void* const* d_in, const int* in_sizes, int n_in,
                              void* d_out, int out_size, void* d_ws, size_t ws_size,
                              hipStream_t stream) {
  const float* cur  = (const float*)d_in[0];
  const float* pre  = (const float*)d_in[1];
  const float* se1w = (const float*)d_in[2];
  const float* se1b = (const float*)d_in[3];
  const float* sw1w = (const float*)d_in[4];
  const float* sw1b = (const float*)d_in[5];
  const float* se2w = (const float*)d_in[6];
  const float* se2b = (const float*)d_in[7];
  const float* sw2w = (const float*)d_in[8];
  const float* sw2b = (const float*)d_in[9];
  const float* se3w = (const float*)d_in[10];
  const float* se3b = (const float*)d_in[11];
  const float* sw3w = (const float*)d_in[12];
  const float* sw3b = (const float*)d_in[13];
  const int*   cidx = (const int*)d_in[14];
  const int*   pidx = (const int*)d_in[15];

  float* dense = (float*)d_ws;
  size_t denseBytes = (size_t)BDIM * FDIM * HWC * CC * sizeof(float);
  hipMemsetAsync(d_ws, 0, denseBytes, stream);

  // Packed weight fragments + bias live just past the dense grid.
  v16h*  bfrag = (v16h*)((char*)d_ws + denseBytes);          // 3*32*2*32 B = 6 KB
  float* bias  = (float*)((char*)d_ws + denseBytes + 8192);  // 3*16 floats

  pack_weights_kernel<<<1, 32, 0, stream>>>(
      se1w, se1b, sw1w, sw1b, se2w, se2b, sw2w, sw2b,
      se3w, se3b, sw3w, sw3b, bfrag, bias);

  int total4 = BDIM * FDIM * MQ * (CC / 4);
  scatter_pre_kernel<<<(total4 + 255) / 256, 256, 0, stream>>>(pre, pidx, dense);

  float* outx     = (float*)d_out;
  float* outspeed = outx + (size_t)BDIM * FDIM * NQ * CC;
  int ntiles = (BDIM * FDIM * NQ) / 16;
  deform_stages_kernel<<<ntiles, 32, 0, stream>>>(
      cur, dense, cidx, bfrag, bias, outx, outspeed);
}
